// OptCssModel_67748814127276
// MI455X (gfx1250) — matrix operand order repaired and checked by reading, not compiled, not run on hardware
//
#include <hip/hip_runtime.h>
#include <cstdint>
#include <cstddef>

// ---------------------------------------------------------------------------
// CSS forward for MI455X (gfx1250, wave32, WMMA).
// catx = x * a_j + c_j   (batchnorm folded to per-column scale/shift)
// h1   = catx @ w1^T + b1
// h1cs = sigmoid(catx @ w2^T + b2)
// out  = clamp(h1) * clamp(h1cs)
//
// GEMMs run as bf16 hi/lo split (3 products) on v_wmma_f32_16x16x32_bf16,
// f32 accumulate. A fragments are shared across both weight matrices.
// ---------------------------------------------------------------------------

typedef __attribute__((ext_vector_type(16))) __bf16 v16bf;
typedef __attribute__((ext_vector_type(4)))  __bf16 v4bf;
typedef __attribute__((ext_vector_type(8)))  float  v8f;

#define N_ROWS 16384
#define K_DIM  2048
#define N_OUT  2048

union FragBF {
  v16bf v;
  uint4 q[2];
};

__device__ __forceinline__ v8f wmma_bf16(v16bf a, v16bf b, v8f c) {
  // (neg_a, A, neg_b, B, c_mod, C, reuse_a, reuse_b)
  return __builtin_amdgcn_wmma_f32_16x16x32_bf16(false, a, false, b,
                                                 (short)0, c, false, false);
}

__device__ __forceinline__ float clamp_small(float v) {
  const float eps = 1e-12f;
  if (v >= 0.0f && v < eps)  return eps;
  if (v < 0.0f && v > -eps)  return -eps;
  return v;
}

struct BfPair { __bf16 hi; __bf16 lo; };

__device__ __forceinline__ BfPair split_bf16(float v) {
  BfPair p;
  p.hi = (__bf16)v;
  p.lo = (__bf16)(v - (float)p.hi);
  return p;
}

// ---------------------------------------------------------------------------
// Stage 1: per-(rowblock, column) partial sums of x and x^2.
// grid = (32 colblocks of 64, 64 rowblocks of 256), block = 256.
// ---------------------------------------------------------------------------
__global__ void css_stats_partial(const float* __restrict__ x,
                                  float* __restrict__ psum,
                                  float* __restrict__ psumsq)
{
  __shared__ float r1[4][64];
  __shared__ float r2[4][64];
  const int cb = blockIdx.x;
  const int rb = blockIdx.y;
  const int cl = threadIdx.x & 63;
  const int rl = threadIdx.x >> 6;      // 0..3
  const int col = cb * 64 + cl;
  const int r0  = rb * 256 + rl;
  float s1 = 0.0f, s2 = 0.0f;
#pragma unroll 4
  for (int i = 0; i < 64; ++i) {
    float v = x[(size_t)(r0 + i * 4) * K_DIM + col];
    s1 += v;
    s2 += v * v;
  }
  r1[rl][cl] = s1;
  r2[rl][cl] = s2;
  __syncthreads();
  if (rl == 0) {
    s1 = r1[0][cl] + r1[1][cl] + r1[2][cl] + r1[3][cl];
    s2 = r2[0][cl] + r2[1][cl] + r2[2][cl] + r2[3][cl];
    psum[rb * K_DIM + col]   = s1;
    psumsq[rb * K_DIM + col] = s2;
  }
}

// ---------------------------------------------------------------------------
// Stage 2: finalize mean / unbiased std, fold into per-column a_j, c_j.
// ---------------------------------------------------------------------------
__global__ void css_stats_final(const float* __restrict__ psum,
                                const float* __restrict__ psumsq,
                                const float* __restrict__ gamma,
                                const float* __restrict__ beta,
                                float* __restrict__ Ac,
                                float* __restrict__ Cc)
{
  int j = blockIdx.x * blockDim.x + threadIdx.x;
  if (j >= K_DIM) return;
  float s1 = 0.0f, s2 = 0.0f;
  for (int rb = 0; rb < 64; ++rb) {
    s1 += psum[rb * K_DIM + j];
    s2 += psumsq[rb * K_DIM + j];
  }
  float mean = s1 * (1.0f / 16384.0f);
  float var  = (s2 - 16384.0f * mean * mean) * (1.0f / 16383.0f); // ddof=1
  float rs   = rsqrtf(var + 1e-8f);
  float a    = rs * gamma[j];
  Ac[j] = a;
  Cc[j] = beta[j] - mean * a;
}

// ---------------------------------------------------------------------------
// Main fused dual-GEMM + epilogue.
// Block tile: M=128, N=64, Kstep=32. 8 waves, wave tile 32x32 per output mat.
// LDS double buffered: A(hi/lo) 128x32 bf16 + B(w1,w2 x hi/lo) 64x32 bf16.
// ---------------------------------------------------------------------------
__global__ __launch_bounds__(256, 1) void css_gemm(
    const float* __restrict__ x,
    const float* __restrict__ w1, const float* __restrict__ b1,
    const float* __restrict__ w2, const float* __restrict__ b2,
    const float* __restrict__ Ac, const float* __restrict__ Cc,
    float* __restrict__ out)
{
  // [stage][row*32 + k]
  __shared__ __bf16 sAh[2][128 * 32];
  __shared__ __bf16 sAl[2][128 * 32];
  // [stage][mat][hi=0/lo=1][n*32 + k]
  __shared__ __bf16 sB[2][2][2][64 * 32];

  const int tid  = threadIdx.x;
  const int lane = tid & 31;
  const int wave = tid >> 5;
  const int mg   = wave & 3;   // 4 wave rows  -> 32 M each
  const int ng   = wave >> 2;  // 2 wave cols  -> 32 N each
  const int rowBase = blockIdx.y * 128;
  const int colBase = blockIdx.x * 64;
  const int hl = lane >> 4;    // lane half
  const int ml = lane & 15;

  v8f acc1[2][2] = {};
  v8f acc2[2][2] = {};

  float4 xa[4];  // staged (already normalized) A values
  float4 wb[4];  // staged raw weight values

  auto stage_load = [&](int k0) {
    // A: 128 rows x 8 float4 chunks = 1024 tasks
#pragma unroll
    for (int i = 0; i < 4; ++i) {
      int ta  = tid + i * 256;
      int row = ta >> 3;
      int k   = k0 + (ta & 7) * 4;
      float4 xv = *(const float4*)&x[(size_t)(rowBase + row) * K_DIM + k];
      float4 av = *(const float4*)&Ac[k];
      float4 cv = *(const float4*)&Cc[k];
      xa[i].x = fmaf(xv.x, av.x, cv.x);
      xa[i].y = fmaf(xv.y, av.y, cv.y);
      xa[i].z = fmaf(xv.z, av.z, cv.z);
      xa[i].w = fmaf(xv.w, av.w, cv.w);
    }
    // B: 2 mats x 64 rows x 8 float4 chunks = 1024 tasks
#pragma unroll
    for (int i = 0; i < 4; ++i) {
      int tb  = tid + i * 256;
      int mat = tb >> 9;
      int rem = tb & 511;
      int n   = rem >> 3;
      int k   = k0 + (rem & 7) * 4;
      const float* w = mat ? w2 : w1;
      wb[i] = *(const float4*)&w[(size_t)(colBase + n) * K_DIM + k];
    }
  };

  auto stage_store = [&](int buf) {
#pragma unroll
    for (int i = 0; i < 4; ++i) {
      int ta  = tid + i * 256;
      int row = ta >> 3;
      int kq  = (ta & 7) * 4;
      BfPair p0 = split_bf16(xa[i].x);
      BfPair p1 = split_bf16(xa[i].y);
      BfPair p2 = split_bf16(xa[i].z);
      BfPair p3 = split_bf16(xa[i].w);
      v4bf hv = { p0.hi, p1.hi, p2.hi, p3.hi };
      v4bf lv = { p0.lo, p1.lo, p2.lo, p3.lo };
      *(v4bf*)&sAh[buf][row * 32 + kq] = hv;
      *(v4bf*)&sAl[buf][row * 32 + kq] = lv;
    }
#pragma unroll
    for (int i = 0; i < 4; ++i) {
      int tb  = tid + i * 256;
      int mat = tb >> 9;
      int rem = tb & 511;
      int n   = rem >> 3;
      int kq  = (rem & 7) * 4;
      BfPair p0 = split_bf16(wb[i].x);
      BfPair p1 = split_bf16(wb[i].y);
      BfPair p2 = split_bf16(wb[i].z);
      BfPair p3 = split_bf16(wb[i].w);
      v4bf hv = { p0.hi, p1.hi, p2.hi, p3.hi };
      v4bf lv = { p0.lo, p1.lo, p2.lo, p3.lo };
      *(v4bf*)&sB[buf][mat][0][n * 32 + kq] = hv;
      *(v4bf*)&sB[buf][mat][1][n * 32 + kq] = lv;
    }
  };

  auto compute = [&](int buf) {
    // A fragments: row m = tile + (lane&15); K chunks at 8h and 8h+16
    FragBF Ah[2], Al[2];
#pragma unroll
    for (int mt = 0; mt < 2; ++mt) {
      int m = mg * 32 + mt * 16 + ml;
      const __bf16* pH = &sAh[buf][m * 32 + hl * 8];
      Ah[mt].q[0] = *(const uint4*)pH;
      Ah[mt].q[1] = *(const uint4*)(pH + 16);
      const __bf16* pL = &sAl[buf][m * 32 + hl * 8];
      Al[mt].q[0] = *(const uint4*)pL;
      Al[mt].q[1] = *(const uint4*)(pL + 16);
    }
#pragma unroll
    for (int nt = 0; nt < 2; ++nt) {
      int n = ng * 32 + nt * 16 + ml;
#pragma unroll
      for (int mat = 0; mat < 2; ++mat) {
        // B fragments: col n; K chunks at 8h and 8h+16
        FragBF Bh, Bl;
        const __bf16* pH = &sB[buf][mat][0][n * 32 + hl * 8];
        Bh.q[0] = *(const uint4*)pH;
        Bh.q[1] = *(const uint4*)(pH + 16);
        const __bf16* pL = &sB[buf][mat][1][n * 32 + hl * 8];
        Bl.q[0] = *(const uint4*)pL;
        Bl.q[1] = *(const uint4*)(pL + 16);
#pragma unroll
        for (int mt = 0; mt < 2; ++mt) {
          v8f& acc = mat ? acc2[mt][nt] : acc1[mt][nt];
          acc = wmma_bf16(Ah[mt].v, Bh.v, acc);   // hi*hi
          acc = wmma_bf16(Ah[mt].v, Bl.v, acc);   // hi*lo
          acc = wmma_bf16(Al[mt].v, Bh.v, acc);   // lo*hi
        }
      }
    }
  };

  // --- pipeline: prologue + 64 K-steps, double buffered ---
  stage_load(0);
  stage_store(0);
  __syncthreads();

  for (int ks = 0; ks < 64; ++ks) {
    const int buf = ks & 1;
    if (ks < 63) stage_load((ks + 1) * 32);
    compute(buf);
    if (ks < 63) stage_store(buf ^ 1);
    __syncthreads();
  }

  // --- fused epilogue: bias, sigmoid, clamp, multiply, store f32 ---
#pragma unroll
  for (int nt = 0; nt < 2; ++nt) {
    const int col = colBase + ng * 32 + nt * 16 + ml;
    const float bb1 = b1[col];
    const float bb2 = b2[col];
#pragma unroll
    for (int mt = 0; mt < 2; ++mt) {
#pragma unroll
      for (int r = 0; r < 8; ++r) {
        int row = rowBase + mg * 32 + mt * 16 + hl * 8 + r;
        float h1 = acc1[mt][nt][r] + bb1;
        float z  = acc2[mt][nt][r] + bb2;
        float sg = 1.0f / (1.0f + expf(-z));
        out[(size_t)row * N_OUT + col] = clamp_small(h1) * clamp_small(sg);
      }
    }
  }
}

// ---------------------------------------------------------------------------
extern "C" void kernel_launch(void* const* d_in, const int* in_sizes, int n_in,
                              void* d_out, int out_size, void* d_ws, size_t ws_size,
                              hipStream_t stream)
{
  (void)in_sizes; (void)n_in; (void)out_size; (void)ws_size;
  const float* x     = (const float*)d_in[0];
  const float* w1    = (const float*)d_in[1];
  const float* b1    = (const float*)d_in[2];
  const float* w2    = (const float*)d_in[3];
  const float* b2    = (const float*)d_in[4];
  const float* gamma = (const float*)d_in[5];
  const float* beta  = (const float*)d_in[6];
  float* out = (float*)d_out;

  float* ws     = (float*)d_ws;
  float* psum   = ws;                    // 64*2048
  float* psumsq = ws + 64 * 2048;        // 64*2048
  float* Ac     = ws + 2 * 64 * 2048;    // 2048
  float* Cc     = Ac + 2048;             // 2048   (~1.06 MB total)

  css_stats_partial<<<dim3(32, 64), 256, 0, stream>>>(x, psum, psumsq);
  css_stats_final<<<dim3(8), 256, 0, stream>>>(psum, psumsq, gamma, beta, Ac, Cc);
  css_gemm<<<dim3(N_OUT / 64, N_ROWS / 128), 256, 0, stream>>>(
      x, w1, b1, w2, b2, Ac, Cc, out);
}
